// XNorConv2D_52922587021943
// MI455X (gfx1250) — matrix operand order repaired and checked by reading, compiled, not run on hardware
//
#include <hip/hip_runtime.h>

typedef __attribute__((ext_vector_type(16))) _Float16 v16h;
typedef __attribute__((ext_vector_type(8)))  _Float16 v8h;
typedef __attribute__((ext_vector_type(8)))  float    v8f;
typedef __attribute__((__vector_size__(4 * sizeof(int)))) int v4i;

#define CIN   256
#define COUT  512
#define HW    256
#define HP    258          // padded spatial
#define KTAPS 9
#define KTOT  2304         // 9 * 256
#define LDA   40           // LDS row stride in halves: 80B -> conflict-free, 16B aligned

// ---- CDNA5 async global->LDS path (guarded; falls back to reg staging) ----
#if defined(__gfx1250__) && __has_builtin(__builtin_amdgcn_global_load_async_to_lds_b128)
#define USE_ASYNC 1
#else
#define USE_ASYNC 0
#endif

__device__ __forceinline__ void async_b128(const void* gsrc, void* ldst) {
#if USE_ASYNC
  // Param types per hipcc diagnostic: (v4i __device__*, v4i __shared__*, imm, imm)
  __builtin_amdgcn_global_load_async_to_lds_b128(
      (__attribute__((address_space(1))) v4i*)gsrc,
      (__attribute__((address_space(3))) v4i*)ldst, 0, 0);
#else
  (void)gsrc; (void)ldst;
#endif
}
__device__ __forceinline__ void wait_async_zero() {
#if USE_ASYNC
#if __has_builtin(__builtin_amdgcn_s_wait_asynccnt)
  __builtin_amdgcn_s_wait_asynccnt(0);
#else
  asm volatile("s_wait_asynccnt 0x0" ::: "memory");
#endif
#endif
}

// ---------------------------------------------------------------------------
// P1: read x (NCHW fp32) once; emit binarized ±1 f16 in padded HWC layout
//     (padding value 1.0 baked in => sign +1, |.|=1) and the channel abs-mean.
// ---------------------------------------------------------------------------
__global__ __launch_bounds__(256) void k_binarize(const float* __restrict__ x,
                                                  _Float16* __restrict__ xq,
                                                  float* __restrict__ absmean) {
  int p = blockIdx.x * 256 + threadIdx.x;
  if (p >= HP * HP) return;
  int hp = p / HP, wp = p % HP;
  bool interior = (hp >= 1) && (hp <= HW) && (wp >= 1) && (wp <= HW);
  const float* xs = x + (size_t)(hp - 1) * HW + (wp - 1);  // only deref'd if interior
  float sum = 0.f;
  _Float16* dst = xq + (size_t)p * CIN;
  for (int c0 = 0; c0 < CIN; c0 += 8) {
    v8h pack;
#pragma unroll
    for (int i = 0; i < 8; ++i) {
      float v = interior ? xs[(size_t)(c0 + i) * (HW * HW)] : 1.0f;
      sum += fabsf(v);
      pack[i] = (v >= 0.f) ? (_Float16)1.0f : (_Float16)(-1.0f);
    }
    *(v8h*)(dst + c0) = pack;   // 16B store, contiguous in channel dim
  }
  absmean[p] = sum * (1.0f / CIN);
}

// ---------------------------------------------------------------------------
// P2: input_K = 3x3 average of abs-mean (VALID over the padded grid).
// ---------------------------------------------------------------------------
__global__ __launch_bounds__(256) void k_kmap(const float* __restrict__ absmean,
                                              float* __restrict__ kmap) {
  int idx = blockIdx.x * 256 + threadIdx.x;
  if (idx >= HW * HW) return;
  int h = idx >> 8, w = idx & 255;
  float s = 0.f;
#pragma unroll
  for (int kh = 0; kh < 3; ++kh)
#pragma unroll
    for (int kw = 0; kw < 3; ++kw)
      s += absmean[(h + kh) * HP + (w + kw)];
  kmap[idx] = s * (1.0f / 9.0f);
}

// ---------------------------------------------------------------------------
// P3: weight -> f16 in [cout][tap*256+cin] layout (K contiguous per row),
//     plus alpha[cout] = mean(weight) via block reduction.
// ---------------------------------------------------------------------------
__global__ __launch_bounds__(256) void k_wxform(const float* __restrict__ w,
                                                _Float16* __restrict__ wt,
                                                float* __restrict__ alpha) {
  __shared__ float red[256];
  int n = blockIdx.x, c = threadIdx.x;
  const float* src = w + (size_t)n * KTOT + (size_t)c * KTAPS;
  float s = 0.f;
#pragma unroll
  for (int tap = 0; tap < KTAPS; ++tap) {
    float v = src[tap];
    s += v;
    wt[(size_t)n * KTOT + tap * CIN + c] = (_Float16)v;
  }
  red[c] = s;
  __syncthreads();
  for (int off = 128; off > 0; off >>= 1) {
    if (c < off) red[c] += red[c + off];
    __syncthreads();
  }
  if (c == 0) alpha[n] = red[0] * (1.0f / KTOT);
}

// ---------------------------------------------------------------------------
// Main implicit-GEMM conv via V_WMMA_F32_16X16X32_F16.
//   GEMM M = cout (512), N = spatial (65536), K = tap-major/cin-minor (2304).
//   Block: 256 threads = 8 wave32 (2x4), tile 128M x 256N x BK=32
//   (one full image row per block). Wave tile 64x64 -> 16 WMMA per 16 ds_b128.
//   Double-buffered LDS; fills go straight to LDS via GLOBAL_LOAD_ASYNC_TO_LDS
//   (ASYNCcnt) when available, else global->reg->ds_store.
// ---------------------------------------------------------------------------
__global__ __launch_bounds__(256) void k_conv_wmma(const _Float16* __restrict__ xq,
                                                   const _Float16* __restrict__ wt,
                                                   const float* __restrict__ kmap,
                                                   const float* __restrict__ alpha,
                                                   float* __restrict__ out) {
  __shared__ __align__(16) _Float16 As[2][128 * LDA];  // weight tile [m][k]  (20 KB)
  __shared__ __align__(16) _Float16 Bs[2][256 * LDA];  // xb tile     [n][k]  (40 KB)
  __shared__ float sAlpha[128];
  __shared__ float sK[256];

  const int tid = threadIdx.x;
  const int h  = blockIdx.x;              // 0..255: one image row per block
  const int n0 = blockIdx.y * 128;        // cout tile base (GEMM M)

  if (tid < 128) sAlpha[tid] = alpha[n0 + tid];
  sK[tid] = kmap[h * HW + tid];

  const int wid    = tid >> 5;
  const int lane   = tid & 31;
  const int wave_m = wid >> 2;            // 0..1 -> 64 cout rows each
  const int wave_n = wid & 3;             // 0..3 -> 64 spatial cols each

  const int arow = tid >> 1;              // 0..127: A tile row this thread stages
  const int asel = tid & 1;               // which 16-half chunk of the 32-K row

  v8f acc[4][4];
#pragma unroll
  for (int i = 0; i < 4; ++i)
#pragma unroll
    for (int j = 0; j < 4; ++j) acc[i][j] = (v8f)0.f;

  const int lr = lane & 15;               // row-within-fragment
  const int hs = lane >> 4;               // lane half

#if USE_ASYNC
  // --- async pipeline: fills bypass VGPRs entirely ---
  auto issue_tile = [&](int t, int buf) {
    int tap  = t >> 3;                    // k = tap*256 + cin ; BK=32 => 8 steps/tap
    int kh   = tap / 3, kw = tap % 3;
    int cin0 = (t & 7) * 32;
    const _Float16* ga = wt + (size_t)(n0 + arow) * KTOT + tap * CIN + cin0 + asel * 16;
    _Float16* la = &As[buf][arow * LDA + asel * 16];
    async_b128(ga, la);
    async_b128(ga + 8, la + 8);
    const _Float16* gb = xq + ((size_t)(h + kh) * HP + (kw + tid)) * CIN + cin0;
    _Float16* lb = &Bs[buf][tid * LDA];
#pragma unroll
    for (int c = 0; c < 4; ++c) async_b128(gb + c * 8, lb + c * 8);
  };

  issue_tile(0, 0);
  wait_async_zero();
  __syncthreads();
#else
  // --- fallback pipeline: global->reg (early) then reg->LDS (late) ---
  uint4 ra0, ra1, rb[4];
  auto load_regs = [&](int t) {
    int tap  = t >> 3;
    int kh   = tap / 3, kw = tap % 3;
    int cin0 = (t & 7) * 32;
    const uint4* pa = (const uint4*)(wt + (size_t)(n0 + arow) * KTOT + tap * CIN + cin0 + asel * 16);
    ra0 = pa[0]; ra1 = pa[1];
    const uint4* pb = (const uint4*)(xq + ((size_t)(h + kh) * HP + (kw + tid)) * CIN + cin0);
#pragma unroll
    for (int c = 0; c < 4; ++c) rb[c] = pb[c];
  };
  auto store_lds = [&](int buf) {
    uint4* da = (uint4*)&As[buf][arow * LDA + asel * 16];
    da[0] = ra0; da[1] = ra1;
    uint4* db = (uint4*)&Bs[buf][tid * LDA];
#pragma unroll
    for (int c = 0; c < 4; ++c) db[c] = rb[c];
  };

  load_regs(0);
  store_lds(0);
  __syncthreads();
#endif

  for (int t = 0; t < 72; ++t) {
    const int cur = t & 1;

#if USE_ASYNC
    if (t + 1 < 72) issue_tile(t + 1, 1 - cur);   // DMA into other buffer
#else
    if (t + 1 < 72) load_regs(t + 1);             // hide global latency
#endif

    // B fragments (K x N): lane holds N=lr, lane-half K pattern {0..7,16..23} / {8..15,24..31}
    v16h bfrag[4];
#pragma unroll
    for (int j = 0; j < 4; ++j) {
      int row = wave_n * 64 + j * 16 + lr;
      v8h lo = *(const v8h*)&Bs[cur][row * LDA + hs * 8];
      v8h hi = *(const v8h*)&Bs[cur][row * LDA + 16 + hs * 8];
      bfrag[j] = __builtin_shufflevector(lo, hi, 0,1,2,3,4,5,6,7,8,9,10,11,12,13,14,15);
    }
#pragma unroll
    for (int i = 0; i < 4; ++i) {
      int row = wave_m * 64 + i * 16 + lr;
      // A fragment (M x K): lane-half K pattern {0..7,16..23} / {8..15,24..31}
      v8h lo = *(const v8h*)&As[cur][row * LDA + hs * 8];
      v8h hi = *(const v8h*)&As[cur][row * LDA + 16 + hs * 8];
      v16h afrag = __builtin_shufflevector(lo, hi, 0,1,2,3,4,5,6,7,8,9,10,11,12,13,14,15);
#pragma unroll
      for (int j = 0; j < 4; ++j)
        acc[i][j] = __builtin_amdgcn_wmma_f32_16x16x32_f16(
            false, afrag, false, bfrag[j], (short)0, acc[i][j], false, false);
    }

#if USE_ASYNC
    wait_async_zero();                    // tile t+1 fully landed in LDS
    __syncthreads();                      // publish; also closes reads of buf[cur]
#else
    __syncthreads();                      // all lanes done reading buf[cur]
    if (t + 1 < 72) {
      store_lds((t + 1) & 1);
      __syncthreads();                    // tile t+1 visible before next compute
    }
#endif
  }

  // Epilogue: out[n][h][w] = acc * input_K[h][w] * alpha[n].
  // C layout: VGPR r, lanes 0-15 -> M=r, lanes 16-31 -> M=8+r; N = lane&15.
  // Each half-wave store is 64B contiguous along w.
#pragma unroll
  for (int i = 0; i < 4; ++i) {
#pragma unroll
    for (int j = 0; j < 4; ++j) {
      int nloc = wave_n * 64 + j * 16 + lr;
      float kv = sK[nloc];
#pragma unroll
      for (int r = 0; r < 8; ++r) {
        int mloc = wave_m * 64 + i * 16 + r + hs * 8;
        float v = acc[i][j][r] * kv * sAlpha[mloc];
        out[(size_t)(n0 + mloc) * (HW * HW) + (size_t)h * HW + nloc] = v;
      }
    }
  }
}

// ---------------------------------------------------------------------------
extern "C" void kernel_launch(void* const* d_in, const int* in_sizes, int n_in,
                              void* d_out, int out_size, void* d_ws, size_t ws_size,
                              hipStream_t stream) {
  const float* x = (const float*)d_in[0];   // [1,256,256,256]
  const float* w = (const float*)d_in[1];   // [512,256,3,3]
  float* out = (float*)d_out;               // [1,512,256,256]

  // Workspace layout (~37 MB total):
  char* ws = (char*)d_ws;
  size_t off = 0;
  auto carve = [&](size_t bytes) -> void* {
    void* p = ws + off;
    off = (off + bytes + 255) & ~(size_t)255;
    return p;
  };
  _Float16* xq      = (_Float16*)carve((size_t)HP * HP * CIN * sizeof(_Float16)); // 34.1 MB
  float*    absmean = (float*)   carve((size_t)HP * HP * sizeof(float));          // 266 KB
  float*    kmap    = (float*)   carve((size_t)HW * HW * sizeof(float));          // 256 KB
  _Float16* wt      = (_Float16*)carve((size_t)COUT * KTOT * sizeof(_Float16));   // 2.36 MB
  float*    alpha   = (float*)   carve((size_t)COUT * sizeof(float));             // 2 KB
  (void)in_sizes; (void)n_in; (void)out_size; (void)ws_size;

  k_binarize<<<(HP * HP + 255) / 256, 256, 0, stream>>>(x, xq, absmean);
  k_kmap<<<(HW * HW) / 256, 256, 0, stream>>>(absmean, kmap);
  k_wxform<<<COUT, 256, 0, stream>>>(w, wt, alpha);
  k_conv_wmma<<<dim3(256, 4), 256, 0, stream>>>(xq, wt, kmap, alpha, out);
}
